// Q2A_76991583748620
// MI455X (gfx1250) — matrix operand order repaired and checked by reading, not compiled, not run on hardware
//
#include <hip/hip_runtime.h>

// ============================================================================
// MI455X / gfx1250 implementation: bf16 WMMA (f32 accum) for every GEMM,
// fused LDS-softmax attention, wave32 shuffle reductions.
// Round 4: fix async builtin arg types — global_load_async_to_lds_b128 takes
// pointers to 4xi32 vectors (global AS1 src, LDS AS3 dst), per the compiler
// diagnostic. Double-buffered A-tile staging via ASYNCcnt retained.
// ============================================================================

typedef __attribute__((ext_vector_type(8)))  float  v8f;
typedef __attribute__((ext_vector_type(16))) __bf16 v16bf;

union BF16x16 { v16bf v; __bf16 b[16]; };

__device__ inline __bf16 f2bf(float x) { return (__bf16)x; }   // native cvt

__device__ inline v8f v8zero() {
  v8f z;
#pragma unroll
  for (int i = 0; i < 8; ++i) z[i] = 0.f;
  return z;
}

__device__ inline v8f wmma_bf16(v16bf a, v16bf b, v8f c) {
  return __builtin_amdgcn_wmma_f32_16x16x32_bf16(false, a, false, b, (short)0, c, false, false);
}

#if __has_builtin(__builtin_amdgcn_global_load_async_to_lds_b128) && \
    __has_builtin(__builtin_amdgcn_s_wait_asynccnt)
#define HAVE_ASYNC_LDS 1
typedef int v4i_async __attribute__((vector_size(16)));
typedef __attribute__((address_space(1))) v4i_async* async_gp_t;
typedef __attribute__((address_space(3))) v4i_async* async_lp_t;
#else
#define HAVE_ASYNC_LDS 0
#endif

// A fragment (16x32 bf16, ISA 7.12.2): lane L holds row (L&15);
// element e -> K = kk + (e&7) + 16*(e>>3) + 8*(L>>4)
// Works for both global and LDS source pointers (row = pointer to k=0).
__device__ inline v16bf load_a_frag(const float* __restrict__ row, int kk, int lhalf) {
  const float* p = row + kk + 8 * lhalf;
  float4 x0 = *(const float4*)(p);
  float4 x1 = *(const float4*)(p + 4);
  float4 y0 = *(const float4*)(p + 16);
  float4 y1 = *(const float4*)(p + 20);
  BF16x16 t;
  t.b[0] = f2bf(x0.x); t.b[1] = f2bf(x0.y); t.b[2]  = f2bf(x0.z); t.b[3]  = f2bf(x0.w);
  t.b[4] = f2bf(x1.x); t.b[5] = f2bf(x1.y); t.b[6]  = f2bf(x1.z); t.b[7]  = f2bf(x1.w);
  t.b[8] = f2bf(y0.x); t.b[9] = f2bf(y0.y); t.b[10] = f2bf(y0.z); t.b[11] = f2bf(y0.w);
  t.b[12]= f2bf(y1.x); t.b[13]= f2bf(y1.y); t.b[14] = f2bf(y1.z); t.b[15] = f2bf(y1.w);
  return t.v;
}

// A fragment sourced from an fp32 LDS strip via scalar reads (post-softmax P)
__device__ inline v16bf load_a_frag_lds(const float* __restrict__ s, int ld,
                                        int kb, int l15, int lhalf) {
  const float* row = s + l15 * ld + kb + 8 * lhalf;
  BF16x16 t;
#pragma unroll
  for (int e = 0; e < 16; ++e) {
    int k = (e & 7) + 16 * (e >> 3);
    t.b[e] = f2bf(row[k]);
  }
  return t.v;
}

// B fragment (32x16 bf16): lane L holds col (L&15); K = kk + 8*(L>>4) + (e&7) + 16*(e>>3).
// Row-K variant: K is contiguous in memory; a lane's 16 K values are two runs of 8 (weight rows / K rows).
__device__ inline v16bf load_b_frag_rowK(const float* __restrict__ rowN, int kk, int lhalf) {
  const float* p = rowN + kk + 8 * lhalf;
  float4 a = *(const float4*)(p);
  float4 b = *(const float4*)(p + 4);
  float4 c = *(const float4*)(p + 16);
  float4 d = *(const float4*)(p + 20);
  BF16x16 t;
  t.b[0] = f2bf(a.x); t.b[1] = f2bf(a.y); t.b[2]  = f2bf(a.z); t.b[3]  = f2bf(a.w);
  t.b[4] = f2bf(b.x); t.b[5] = f2bf(b.y); t.b[6]  = f2bf(b.z); t.b[7]  = f2bf(b.w);
  t.b[8] = f2bf(c.x); t.b[9] = f2bf(c.y); t.b[10] = f2bf(c.z); t.b[11] = f2bf(c.w);
  t.b[12]= f2bf(d.x); t.b[13]= f2bf(d.y); t.b[14] = f2bf(d.z); t.b[15] = f2bf(d.w);
  return t.v;
}

// Strided variant: B stored (K,N) row-major (K stride = ldb)
__device__ inline v16bf load_b_frag_strided(const float* __restrict__ colN, int kk,
                                            int lhalf, int ldb) {
  const float* p = colN + (long long)(kk + 8 * lhalf) * ldb;
  BF16x16 t;
#pragma unroll
  for (int e = 0; e < 16; ++e) t.b[e] = f2bf(p[(long long)((e & 7) + 16 * (e >> 3)) * ldb]);
  return t.v;
}

// ============================================================================
// Batched GEMM:  C = act( A(M,K) * B + bias ),
//   bKN==0: B is weights (N,K) row-major (x @ W^T)
//   bKN==1: B is (K,N) row-major (plain A@B)
// alphap!=null -> leaky-relu with device-resident alpha.
// Block: 8 waves (2 M x 4 N), wave tile 32x64, block tile 64x256.
// A tile (64x32 fp32) staged per k-step into double-buffered LDS via
// ASYNC global->LDS copies (ASYNCcnt); per k-step 8 WMMAs per wave.
// ============================================================================
__global__ __launch_bounds__(256) void k_gemm(
    const float* __restrict__ A, const float* __restrict__ Bw,
    const float* __restrict__ bias, const float* __restrict__ alphap,
    float* __restrict__ C,
    int M, int N, int K, int lda, int ldb, int ldc, int bKN,
    long long sA, long long sB, long long sC)
{
  __shared__ float tileA[2][64][32];           // 16 KB, double-buffered

  const float* Ab = A  + (long long)blockIdx.z * sA;
  const float* Bb = Bw + (long long)blockIdx.z * sB;
  float*       Cb = C  + (long long)blockIdx.z * sC;

  int tid   = threadIdx.x;
  int wave  = tid >> 5;
  int lane  = tid & 31;
  int lhalf = lane >> 4;
  int l15   = lane & 15;
  int m0 = blockIdx.y * 64  + (wave & 1) * 32;
  int n0 = blockIdx.x * 256 + (wave >> 1) * 64;
  int m0blk = blockIdx.y * 64;

  // stage k-step [kk, kk+32) of the block's 64 A rows into tileA[buf]
  // 512 float4 slots; thread t handles slots t and t+256.
  auto stage = [&](int buf, int kk) {
#pragma unroll
    for (int i = 0; i < 2; ++i) {
      int s   = tid + i * 256;
      int row = s >> 3;
      int c4  = (s & 7) * 4;
      int gr  = m0blk + row; if (gr > M - 1) gr = M - 1;
      const float* g = Ab + (long long)gr * lda + kk + c4;
      float* l = &tileA[buf][row][c4];
#if HAVE_ASYNC_LDS
      __builtin_amdgcn_global_load_async_to_lds_b128(
          (async_gp_t)(v4i_async*)(void*)g,
          (async_lp_t)(v4i_async*)(void*)l, 0, 0);
#else
      *(float4*)l = *(const float4*)g;
#endif
    }
  };

  v8f acc0[4], acc1[4];
#pragma unroll
  for (int t = 0; t < 4; ++t) { acc0[t] = v8zero(); acc1[t] = v8zero(); }

  int ncol[4];
#pragma unroll
  for (int t = 0; t < 4; ++t) {
    int n = n0 + 16 * t + l15;
    ncol[t] = (n > N - 1) ? (N - 1) : n;
  }

  stage(0, 0);                                 // prime first buffer

  for (int kk = 0; kk < K; kk += 32) {
    int p = (kk >> 5) & 1;
    if (kk + 32 < K) {
      stage(p ^ 1, kk + 32);                   // overlap next copy with compute
#if HAVE_ASYNC_LDS
      __builtin_amdgcn_s_wait_asynccnt(2);     // in-order: current tile landed
#endif
    } else {
#if HAVE_ASYNC_LDS
      __builtin_amdgcn_s_wait_asynccnt(0);
#endif
    }
    __syncthreads();                           // tile[p] valid for all waves

    const float* t0 = &tileA[p][(wave & 1) * 32 + l15][0];
    v16bf a0 = load_a_frag(t0, 0, lhalf);
    v16bf a1 = load_a_frag(t0 + 16 * 32, 0, lhalf);
    if (bKN == 0 && kk + 32 < K)
      __builtin_prefetch(Bb + (long long)ncol[0] * ldb + kk + 32, 0, 1);
#pragma unroll
    for (int t = 0; t < 4; ++t) {
      v16bf bf;
      if (bKN == 0) bf = load_b_frag_rowK(Bb + (long long)ncol[t] * ldb, kk, lhalf);
      else          bf = load_b_frag_strided(Bb + ncol[t], kk, lhalf, ldb);
      acc0[t] = wmma_bf16(a0, bf, acc0[t]);
      acc1[t] = wmma_bf16(a1, bf, acc1[t]);
    }
    __syncthreads();                           // tile[p] free for re-staging
  }

  float alpha = alphap ? alphap[0] : 0.f;
  bool  lact  = (alphap != nullptr);
#pragma unroll
  for (int t = 0; t < 4; ++t) {
    int nc = n0 + 16 * t + l15;
    if (nc >= N) continue;
    float bv = bias ? bias[nc] : 0.f;
#pragma unroll
    for (int r = 0; r < 8; ++r) {              // C layout: m = r + 8*lhalf
      int mrA = m0 + r + 8 * lhalf;
      if (mrA < M) {
        float vv = acc0[t][r] + bv;
        if (lact && vv < 0.f) vv *= alpha;
        Cb[(long long)mrA * ldc + nc] = vv;
      }
      int mrB = mrA + 16;
      if (mrB < M) {
        float vv = acc1[t][r] + bv;
        if (lact && vv < 0.f) vv *= alpha;
        Cb[(long long)mrB * ldc + nc] = vv;
      }
    }
  }
}

// ============================================================================
// Fused attention (output mode). Heads interleaved in D: head h = cols [64h,64h+64).
// Grid: (Lq/16, NH=16, nbatch). Scores strip (16 x Lk<=1024) lives in LDS.
// ============================================================================
__global__ __launch_bounds__(256) void k_attn_o(
    const float* __restrict__ Qp, const float* __restrict__ Kp,
    const float* __restrict__ Vp, float* __restrict__ Op,
    int Lq, int Lk, int kvdiv, long long sQ, long long sKV, float scale)
{
  __shared__ float sc[16][1024];               // 64 KB
  int b = blockIdx.z, h = blockIdx.y, qt = blockIdx.x;
  const float* Qb = Qp + (long long)b * sQ + h * 64;
  const float* Kb = Kp + (long long)(b / kvdiv) * sKV + h * 64;
  const float* Vb = Vp + (long long)(b / kvdiv) * sKV + h * 64;
  float*       Ob = Op + (long long)b * sQ + h * 64;

  int wave = threadIdx.x >> 5, lane = threadIdx.x & 31;
  int lhalf = lane >> 4, l15 = lane & 15;
  int q0 = qt * 16;

  const float* Qrow = Qb + (long long)(q0 + l15) * 1024;
  v16bf aq0 = load_a_frag(Qrow, 0, lhalf);     // hd 0..31
  v16bf aq1 = load_a_frag(Qrow, 32, lhalf);    // hd 32..63
  int nchunk = Lk >> 5;

  // Phase 1: S = Q K^T * scale  -> LDS
  for (int c = wave; c < nchunk; c += 8) {
    int kb = c * 32;
    const float* K0 = Kb + (long long)(kb + l15) * 1024;
    const float* K1 = Kb + (long long)(kb + 16 + l15) * 1024;
    v8f s0 = v8zero(), s1 = v8zero();
    s0 = wmma_bf16(aq0, load_b_frag_rowK(K0, 0, lhalf), s0);
    s0 = wmma_bf16(aq1, load_b_frag_rowK(K0, 32, lhalf), s0);
    s1 = wmma_bf16(aq0, load_b_frag_rowK(K1, 0, lhalf), s1);
    s1 = wmma_bf16(aq1, load_b_frag_rowK(K1, 32, lhalf), s1);
#pragma unroll
    for (int r = 0; r < 8; ++r) {
      sc[r + 8 * lhalf][kb + l15]      = s0[r] * scale;
      sc[r + 8 * lhalf][kb + 16 + l15] = s1[r] * scale;
    }
  }
  __syncthreads();

  // Phase 2: row softmax (16 threads per row, shuffle reductions width 16)
  {
    int row = threadIdx.x >> 4, sub = threadIdx.x & 15;
    float mx = -1e30f;
    for (int k = sub; k < Lk; k += 16) mx = fmaxf(mx, sc[row][k]);
#pragma unroll
    for (int m = 1; m < 16; m <<= 1) mx = fmaxf(mx, __shfl_xor(mx, m, 16));
    float sum = 0.f;
    for (int k = sub; k < Lk; k += 16) { float e = __expf(sc[row][k] - mx); sc[row][k] = e; sum += e; }
#pragma unroll
    for (int m = 1; m < 16; m <<= 1) sum += __shfl_xor(sum, m, 16);
    float inv = 1.f / sum;
    for (int k = sub; k < Lk; k += 16) sc[row][k] *= inv;
  }
  __syncthreads();

  // Phase 3: O = P V   (waves 0..3 each own 16 hd cols)
  if (wave < 4) {
    v8f o = v8zero();
    for (int c = 0; c < nchunk; ++c) {
      int kb = c * 32;
      v16bf ap = load_a_frag_lds(&sc[0][0], 1024, kb, l15, lhalf);
      v16bf bv = load_b_frag_strided(Vb + (long long)kb * 1024 + 16 * wave + l15, 0, lhalf, 1024);
      o = wmma_bf16(ap, bv, o);
    }
#pragma unroll
    for (int r = 0; r < 8; ++r)
      Ob[(long long)(q0 + r + 8 * lhalf) * 1024 + 16 * wave + l15] = o[r];
  }
}

// ============================================================================
// Attention probability mode for qa2s: w = mean_h softmax(Q K^T / sqrt(hd)).
// Loops all 16 heads inside the workgroup -> deterministic, no atomics.
// Lk fixed at 512. Grid: (Lq/16, 1, nbatch).
// ============================================================================
__global__ __launch_bounds__(256) void k_attn_w(
    const float* __restrict__ Qp, const float* __restrict__ Kp,
    float* __restrict__ Wout, int kvdiv,
    long long sQ, long long sK, long long sW, float scale)
{
  __shared__ float sc[16][512];
  __shared__ float wacc[16][512];              // 64 KB total
  const int Lk = 512;
  int b = blockIdx.z, qt = blockIdx.x;
  int wave = threadIdx.x >> 5, lane = threadIdx.x & 31;
  int lhalf = lane >> 4, l15 = lane & 15;
  int q0 = qt * 16;

  for (int i = threadIdx.x; i < 16 * 512; i += 256) (&wacc[0][0])[i] = 0.f;
  __syncthreads();

  for (int h = 0; h < 16; ++h) {
    const float* Qb = Qp + (long long)b * sQ + h * 64;
    const float* Kb = Kp + (long long)(b / kvdiv) * sK + h * 64;
    const float* Qrow = Qb + (long long)(q0 + l15) * 1024;
    v16bf aq0 = load_a_frag(Qrow, 0, lhalf);
    v16bf aq1 = load_a_frag(Qrow, 32, lhalf);
    for (int c = wave; c < (Lk >> 5); c += 8) {
      int kb = c * 32;
      const float* K0 = Kb + (long long)(kb + l15) * 1024;
      const float* K1 = Kb + (long long)(kb + 16 + l15) * 1024;
      v8f s0 = v8zero(), s1 = v8zero();
      s0 = wmma_bf16(aq0, load_b_frag_rowK(K0, 0, lhalf), s0);
      s0 = wmma_bf16(aq1, load_b_frag_rowK(K0, 32, lhalf), s0);
      s1 = wmma_bf16(aq0, load_b_frag_rowK(K1, 0, lhalf), s1);
      s1 = wmma_bf16(aq1, load_b_frag_rowK(K1, 32, lhalf), s1);
#pragma unroll
      for (int r = 0; r < 8; ++r) {
        sc[r + 8 * lhalf][kb + l15]      = s0[r] * scale;
        sc[r + 8 * lhalf][kb + 16 + l15] = s1[r] * scale;
      }
    }
    __syncthreads();
    {
      int row = threadIdx.x >> 4, sub = threadIdx.x & 15;
      float mx = -1e30f;
      for (int k = sub; k < Lk; k += 16) mx = fmaxf(mx, sc[row][k]);
#pragma unroll
      for (int m = 1; m < 16; m <<= 1) mx = fmaxf(mx, __shfl_xor(mx, m, 16));
      float sum = 0.f;
      for (int k = sub; k < Lk; k += 16) { float e = __expf(sc[row][k] - mx); sc[row][k] = e; sum += e; }
#pragma unroll
      for (int m = 1; m < 16; m <<= 1) sum += __shfl_xor(sum, m, 16);
      float inv = 1.f / sum;
      for (int k = sub; k < Lk; k += 16) wacc[row][k] += sc[row][k] * inv * 0.0625f;
    }
    __syncthreads();
  }
  for (int i = threadIdx.x; i < 16 * 512; i += 256) {
    int r = i >> 9, k = i & 511;
    Wout[(long long)b * sW + (long long)(q0 + r) * 512 + k] = wacc[r][k];
  }
}

// ======================= small helper kernels ===============================
__global__ void k_concat(const float* __restrict__ a, const float* __restrict__ b,
                         const float* __restrict__ c, const float* __restrict__ d,
                         float* __restrict__ out, int n) {
  int i = blockIdx.x * 256 + threadIdx.x;
  if (i >= n) return;
  int row = i >> 12, col = i & 4095;
  const float* src = (col < 1024) ? a : (col < 2048) ? b : (col < 3072) ? c : d;
  out[i] = src[row * 1024 + (col & 1023)];
}

__global__ void k_gh(const float* __restrict__ state, const float* __restrict__ wh,
                     const float* __restrict__ bh, float* __restrict__ gh) {
  int j = blockIdx.x * 256 + threadIdx.x;
  if (j >= 3072) return;
  float sum = bh[j];
  const float* w = wh + (long long)j * 1024;
  for (int k = 0; k < 1024; ++k) sum += state[k] * w[k];
  gh[j] = sum;
}

__global__ void k_gru(const float* __restrict__ gx, const float* __restrict__ gh,
                      const float* __restrict__ state, float* __restrict__ out, int n) {
  int i = blockIdx.x * 256 + threadIdx.x;
  if (i >= n) return;
  int row = i >> 10, j = i & 1023;
  const float* g = gx + (long long)row * 3072;
  float r  = 1.f / (1.f + __expf(-(g[j] + gh[j])));
  float z  = 1.f / (1.f + __expf(-(g[1024 + j] + gh[1024 + j])));
  float nn = tanhf(g[2048 + j] + r * gh[2048 + j]);
  out[i] = (1.f - z) * nn + z * state[j];
}

__global__ void k_ce(const float* __restrict__ logits, const int* __restrict__ labels,
                     float* __restrict__ out) {
  __shared__ float red[128];
  int t = threadIdx.x;                         // 128 = B*S rows, A=32 each
  const float* x = logits + t * 32;
  float m = x[0];
  for (int a = 1; a < 32; ++a) m = fmaxf(m, x[a]);
  float s = 0.f;
  for (int a = 0; a < 32; ++a) s += __expf(x[a] - m);
  float lse = __logf(s) + m;
  red[t] = lse - x[labels[t]];
  __syncthreads();
  for (int st = 64; st > 0; st >>= 1) { if (t < st) red[t] += red[t + st]; __syncthreads(); }
  if (t == 0) out[0] = red[0] / 128.f;
}

// ============================================================================
// Host orchestration
// ============================================================================
extern "C" void kernel_launch(void* const* d_in, const int* in_sizes, int n_in,
                              void* d_out, int out_size, void* d_ws, size_t ws_size,
                              hipStream_t stream) {
  (void)in_sizes; (void)n_in; (void)out_size; (void)ws_size;

  // Assumed input order: top-level dict in insertion order,
  // 'params' flattened as a JAX pytree (alphabetical keys at every level):
  //   0 video  1 script  2 question  3 a_texts  4 a_buttons  5 labels
  //   params: ab2at(6) gru(14) mlp_pre(18) mlp_t(26) mlp_v(34) proj(42)
  //           q2a(50) qa2s(58) s2v(66) state(74)
  //   mha leaves: bk,bo,bq,bv,wk,wo,wq,wv     mlp leaves: a1,a2,b1,b2,b3,w1,w2,w3
  //   gru leaves: bh,bi,wh,wi
  auto f = [&](int i) { return (const float*)d_in[i]; };
  enum { P_AB2AT = 6, P_GRU = 14, P_MLP_PRE = 18, P_MLP_T = 26, P_MLP_V = 34,
         P_PROJ = 42, P_Q2A = 50, P_QA2S = 58, P_S2V = 66, P_STATE = 74 };

  float* W = (float*)d_ws;
  size_t off = 0;
  auto alloc = [&](size_t n) { float* p = W + off; off += (n + 63) & ~(size_t)63; return p; };

  float* AR_A = alloc(16384 * 1024);
  float* AR_B = alloc(16384 * 1024);
  float* AR_C = alloc(16384 * 1024);
  float* VBUF = alloc(16384 * 1024);   // v, later reused for concat (4096x4096)
  float* SBUF = alloc(8192 * 1024);
  float* VIDB = alloc(8192 * 1024);
  float* QBUF = alloc(512 * 1024);
  float* ATB  = alloc(4096 * 1024);
  float* ABB  = alloc(4096 * 1024);
  float* ANSB = alloc(4096 * 1024);
  float* Q2AB = alloc(4096 * 1024);
  float* WB   = alloc(16 * 8 * 32 * 512);
  float* QAVB = alloc(4096 * 1024);
  float* INPB = alloc(4096 * 1024);
  float* STB  = alloc(4096 * 1024);
  float* GHB  = alloc(3072);
  float* LOGB = alloc(4096);

  auto gemm = [&](const float* A, const float* Bw, const float* bias, const float* alphap,
                  float* C, int M, int N, int K, int ldb, int ldc, int bKN,
                  int nb, long long sA, long long sB, long long sC) {
    dim3 g((N + 255) / 256, (M + 63) / 64, nb);
    k_gemm<<<g, dim3(256, 1, 1), 0, stream>>>(A, Bw, bias, alphap, C,
                                              M, N, K, K, ldb, ldc, bKN, sA, sB, sC);
  };

  // mlp leaves: a1+0 a2+1 b1+2 b2+3 b3+4 w1+5 w2+6 w3+7
  auto mlp = [&](const float* X, int M, int din, int dout, int ib,
                 float* t1, float* t2, float* out) {
    gemm(X,  f(ib + 5), f(ib + 2), f(ib + 0), t1, M, din, din, din, din, 0, 1, 0, 0, 0);
    gemm(t1, f(ib + 6), f(ib + 3), f(ib + 1), t2, M, din, din, din, din, 0, 1, 0, 0, 0);
    gemm(t2, f(ib + 7), f(ib + 4), nullptr,  out, M, dout, din, din, dout, 0, 1, 0, 0, 0);
  };

  // mha leaves: bk+0 bo+1 bq+2 bv+3 wk+4 wo+5 wq+6 wv+7
  auto mha = [&](const float* Xq, int Mq, const float* Xkv, int Mkv, int ib,
                 int Lq, int Lk, int nb, int kvdiv,
                 float* qp, float* kp, float* vp, float* op, float* out) {
    gemm(Xq,  f(ib + 6), f(ib + 2), nullptr, qp, Mq,  1024, 1024, 1024, 1024, 0, 1, 0, 0, 0);
    gemm(Xkv, f(ib + 4), f(ib + 0), nullptr, kp, Mkv, 1024, 1024, 1024, 1024, 0, 1, 0, 0, 0);
    gemm(Xkv, f(ib + 7), f(ib + 3), nullptr, vp, Mkv, 1024, 1024, 1024, 1024, 0, 1, 0, 0, 0);
    dim3 g(Lq / 16, 16, nb);
    k_attn_o<<<g, 256, 0, stream>>>(qp, kp, vp, op, Lq, Lk, kvdiv,
                                    (long long)Lq * 1024, (long long)Lk * 1024, 0.125f);
    gemm(op, f(ib + 5), f(ib + 1), nullptr, out, Mq, 1024, 1024, 1024, 1024, 0, 1, 0, 0, 0);
  };

  // 1. v = mlp_v(video)            (16384 rows)
  mlp(f(0), 16384, 1024, 1024, P_MLP_V, AR_A, AR_B, VBUF);
  // 2. s = mlp_t(script)           (8192 rows)
  mlp(f(1), 8192, 1024, 1024, P_MLP_T, AR_A, AR_B, SBUF);
  // 3. vid = MHA(s, v, v)          b=16, Lq=512, Lk=1024
  mha(SBUF, 8192, VBUF, 16384, P_S2V, 512, 1024, 16, 1,
      AR_A, AR_B, AR_C, AR_A + (size_t)8192 * 1024, VIDB);
  // 4. q = mlp_t(question)
  mlp(f(2), 512, 1024, 1024, P_MLP_T, AR_A, AR_B, QBUF);
  // 5. at = mlp_t(a_texts), 6. ab = mlp_v(a_buttons)
  mlp(f(3), 4096, 1024, 1024, P_MLP_T, AR_A, AR_B, ATB);
  mlp(f(4), 4096, 1024, 1024, P_MLP_V, AR_A, AR_B, ABB);
  // 7. ans = MHA(ab, at, at)       batch=(b,s)=128, Lq=Lk=32
  mha(ABB, 4096, ATB, 4096, P_AB2AT, 32, 32, 128, 1,
      AR_A, AR_B, AR_C, AR_A + (size_t)4096 * 1024, ANSB);
  // 8. q2a = MHA(ans, q, q)        kv broadcast over s (kvdiv=8)
  mha(ANSB, 4096, QBUF, 512, P_Q2A, 32, 32, 128, 8,
      AR_A, AR_B, AR_C, AR_A + (size_t)4096 * 1024, Q2AB);
  // 9. w = mean_h softmax(q2a,s)   probabilities only
  gemm(Q2AB, f(P_QA2S + 6), f(P_QA2S + 2), nullptr, AR_A, 4096, 1024, 1024, 1024, 1024, 0, 1, 0, 0, 0);
  gemm(SBUF, f(P_QA2S + 4), f(P_QA2S + 0), nullptr, AR_B, 8192, 1024, 1024, 1024, 1024, 0, 1, 0, 0, 0);
  k_attn_w<<<dim3(2, 1, 128), 256, 0, stream>>>(AR_A, AR_B, WB, 8,
      32LL * 1024, 512LL * 1024, 32LL * 512, 0.125f);
  // 10. qa_v = w @ vid             batched, B is (K,N)
  gemm(WB, VIDB, nullptr, nullptr, QAVB, 256, 1024, 512, 1024, 1024, 1,
       16, 256LL * 512, 512LL * 1024, 256LL * 1024);
  // 11. concat -> (4096, 4096)
  k_concat<<<(16777216 + 255) / 256, 256, 0, stream>>>(QAVB, Q2AB, ABB, ATB, VBUF, 16777216);
  // 12. inp = mlp_pre(concat)
  mlp(VBUF, 4096, 4096, 1024, P_MLP_PRE, AR_A, AR_B, INPB);
  // 13. gx = inp @ wi^T + bi       (gru leaves: bh+0 bi+1 wh+2 wi+3)
  gemm(INPB, f(P_GRU + 3), f(P_GRU + 1), nullptr, AR_A, 4096, 3072, 1024, 1024, 3072, 0, 1, 0, 0, 0);
  // 14. gh = state @ wh^T + bh     (h0 is broadcast -> single vector)
  k_gh<<<12, 256, 0, stream>>>(f(P_STATE), f(P_GRU + 2), f(P_GRU + 0), GHB);
  // 15. states = GRU cell
  k_gru<<<(4096 * 1024 + 255) / 256, 256, 0, stream>>>(AR_A, GHB, f(P_STATE), STB, 4096 * 1024);
  // 16. logits = proj(states)      (N=1 final layer)
  mlp(STB, 4096, 1024, 1, P_PROJ, AR_B, AR_C, LOGB);
  // 17. mean CE
  k_ce<<<1, 128, 0, stream>>>(LOGB, (const int*)d_in[5], (float*)d_out);
}
